// TetralemmaAttention_67800353735172
// MI455X (gfx1250) — compile-verified, matrix operand order checked by reading, not run on hardware
//
#include <hip/hip_runtime.h>

// ---------------------------------------------------------------------------
// TetralemmaAttention for MI455X (gfx1250, wave32, WMMA + TDM)
// B=2, S=2048, H=1024, NH=16, HD=64  ->  M = B*S = 4096
// ---------------------------------------------------------------------------

typedef __attribute__((ext_vector_type(16))) __bf16        v16bf;
typedef __attribute__((ext_vector_type(8)))  float         v8f;
typedef __attribute__((ext_vector_type(4)))  unsigned int  v4u;
typedef __attribute__((ext_vector_type(8)))  int           v8i;
typedef __attribute__((ext_vector_type(4)))  int           v4i;

#define WMMA_BF16(A, B, C) \
  __builtin_amdgcn_wmma_f32_16x16x32_bf16(false, (A), false, (B), (short)0, (C), false, false)

__device__ __forceinline__ __bf16 f2bf(float f) { return (__bf16)f; }

// 32B load through a possibly-16B-aligned pointer (LDS fragment rows are
// 80B-strided for bank-conflict avoidance, so only 16B alignment holds).
__device__ __forceinline__ v16bf frag_ld(const __bf16* p) {
  v16bf r;
  __builtin_memcpy(&r, p, 32);
  return r;
}

#if defined(__AMDGCN__) && __has_builtin(__builtin_amdgcn_tensor_load_to_lds)
#define USE_TDM 1
#else
#define USE_TDM 0
#endif

#if USE_TDM
// Issue one TDM load of a (rows=128) x (cols=32) bf16 tile into LDS.
// Tensor in memory: W[N][K] bf16 row-major; tile starts at (row0, kc).
// LDS padding: +4 DWORDs after every 16 DWORDs (64B row) -> 80B row stride.
__device__ __forceinline__ void tdm_load_wtile(const __bf16* W, int N, int K,
                                               int row0, int kc,
                                               unsigned lds_off) {
  unsigned long long ga =
      (unsigned long long)(uintptr_t)(W + (size_t)row0 * K + kc);
  v4u g0;
  g0[0] = 1u;                                            // count=1, no gather
  g0[1] = lds_off;                                       // LDS byte address
  g0[2] = (unsigned)(ga & 0xFFFFFFFFu);                  // global_addr[31:0]
  g0[3] = (unsigned)((ga >> 32) & 0x01FFFFFFu) | 0x80000000u;  // addr[56:32] | type=2

  v8i g1;
  g1[0] = (1 << 16)        // data_size = 1 (2 bytes)
        | (1 << 20)        // pad_enable
        | (3 << 22)        // pad_interval code 3 = 16 DWORDs (64B)
        | (3 << 25);       // pad_amount  code 3 = 4 DWORDs (16B)
  g1[1] = (K & 0xFFFF) << 16;                            // tensor_dim0[15:0]
  g1[2] = ((K >> 16) & 0xFFFF) | ((N & 0xFFFF) << 16);   // dim0 hi | dim1 lo
  g1[3] = ((N >> 16) & 0xFFFF) | (32 << 16);             // dim1 hi | tile_dim0=32
  g1[4] = 128;                                           // tile_dim1=128, tile_dim2=0
  g1[5] = K;                                             // tensor_dim0_stride lo
  g1[6] = 0;                                             // stride0 hi | stride1 lo
  g1[7] = 0;

  v4i z4 = {0, 0, 0, 0};
#if __clang_major__ >= 23
  v8i z8 = {0, 0, 0, 0, 0, 0, 0, 0};
  __builtin_amdgcn_tensor_load_to_lds(g0, g1, z4, z4, z8, 0);
#else
  __builtin_amdgcn_tensor_load_to_lds(g0, g1, z4, z4, 0);
#endif
}
#endif

// ---------------------------------------------------------------------------
// GEMM: C[M,N] = A[M,K] @ W[N,K]^T + bias[N]; A,W bf16 in memory, f32 acc,
// f32 C. 256 threads = 8 waves, WG tile 64Mx128N, per-wave 32x32.
// W tile (128x32 bf16) is staged through LDS by the Tensor Data Mover,
// double-buffered, issued by wave 0, synchronized with TENSORcnt + barrier.
// A fragments are direct 32B global loads (L2-resident working set).
// ---------------------------------------------------------------------------
#define WROW 80   // LDS row stride in bytes (64B data + 16B TDM pad)

__global__ __launch_bounds__(256) void gemm_bf16_kernel(
    const __bf16* __restrict__ A, const __bf16* __restrict__ W,
    const float* __restrict__ bias, float* __restrict__ C,
    int M, int N, int K) {
  const int lane  = threadIdx.x & 31;
  const int wave  = threadIdx.x >> 5;
  const int lrow  = lane & 15;
  const int lhalf = lane >> 4;
  const int wm = wave >> 2;               // 0..1
  const int wn = wave & 3;                // 0..3
  const int m0 = blockIdx.x * 64 + wm * 32;
  const int n0 = blockIdx.y * 128 + wn * 32;

  v8f acc00 = {}, acc01 = {}, acc10 = {}, acc11 = {};
  const __bf16* a0p = A + (size_t)(m0 + lrow) * K + lhalf * 16;
  const __bf16* a1p = A + (size_t)(m0 + 16 + lrow) * K + lhalf * 16;

#if USE_TDM
  __shared__ __align__(128) char Wl[2][128 * WROW];
  const bool issuer = (threadIdx.x < 32);  // wave 0 issues DMA (wave-uniform)
  const char* f0 = &Wl[0][(wn * 32 + lrow) * WROW + lhalf * 32];
  const char* f1 = f0 + 16 * WROW;

  if (issuer) {
    tdm_load_wtile(W, N, K, blockIdx.y * 128, 0, (unsigned)(uintptr_t)&Wl[0][0]);
    __builtin_amdgcn_s_wait_tensorcnt(0);
  }
  __syncthreads();

  int buf = 0;
  for (int kc = 0; kc < K; kc += 32) {
    const int nkc = kc + 32;
    if (nkc < K && issuer)
      tdm_load_wtile(W, N, K, blockIdx.y * 128, nkc,
                     (unsigned)(uintptr_t)&Wl[buf ^ 1][0]);

    const v16bf a0 = *(const v16bf*)(a0p + kc);
    const v16bf a1 = *(const v16bf*)(a1p + kc);
    const v16bf b0 = frag_ld((const __bf16*)(f0 + buf * (128 * WROW)));
    const v16bf b1 = frag_ld((const __bf16*)(f1 + buf * (128 * WROW)));
    acc00 = WMMA_BF16(a0, b0, acc00);
    acc01 = WMMA_BF16(a0, b1, acc01);
    acc10 = WMMA_BF16(a1, b0, acc10);
    acc11 = WMMA_BF16(a1, b1, acc11);

    if (nkc < K) {
      if (issuer) __builtin_amdgcn_s_wait_tensorcnt(0);
      __syncthreads();         // next buffer ready; cur buffer free to refill
      buf ^= 1;
    }
  }
#else
  const __bf16* w0p = W + (size_t)(n0 + lrow) * K + lhalf * 16;
  const __bf16* w1p = W + (size_t)(n0 + 16 + lrow) * K + lhalf * 16;
  for (int kc = 0; kc < K; kc += 32) {
    const v16bf a0 = *(const v16bf*)(a0p + kc);
    const v16bf a1 = *(const v16bf*)(a1p + kc);
    const v16bf b0 = *(const v16bf*)(w0p + kc);
    const v16bf b1 = *(const v16bf*)(w1p + kc);
    acc00 = WMMA_BF16(a0, b0, acc00);
    acc01 = WMMA_BF16(a0, b1, acc01);
    acc10 = WMMA_BF16(a1, b0, acc10);
    acc11 = WMMA_BF16(a1, b1, acc11);
  }
#endif

  const float bv0 = bias[n0 + lrow];
  const float bv1 = bias[n0 + 16 + lrow];
  // C layout: VGPR r, lane -> row = r + 8*lhalf, col = lrow (per 16x16 tile)
#pragma unroll
  for (int r = 0; r < 8; ++r) {
    const int row0 = m0 + r + 8 * lhalf;
    const int row1 = m0 + 16 + r + 8 * lhalf;
    C[(size_t)row0 * N + n0 + lrow]      = acc00[r] + bv0;
    C[(size_t)row0 * N + n0 + 16 + lrow] = acc01[r] + bv1;
    C[(size_t)row1 * N + n0 + lrow]      = acc10[r] + bv0;
    C[(size_t)row1 * N + n0 + 16 + lrow] = acc11[r] + bv1;
  }
}

// f32 -> bf16 elementwise (memory-bound one-shot pass)
__global__ void cvt_bf16_kernel(const float* __restrict__ in,
                                __bf16* __restrict__ out, int n) {
  const int i = blockIdx.x * blockDim.x + threadIdx.x;
  if (i < n) out[i] = f2bf(in[i]);
}

// ---------------------------------------------------------------------------
// Pack the 64 needed Q/K projection rows (head-dim slots 0 and 2 only) into a
// 128x1024 bf16 weight (rows 64..127 zero) + f32 bias.
// ---------------------------------------------------------------------------
__global__ void pack_qk_kernel(const float* __restrict__ Wq,
                               const float* __restrict__ Wk,
                               const float* __restrict__ bq,
                               const float* __restrict__ bk,
                               __bf16* __restrict__ Wp, float* __restrict__ bp) {
  const int tid = blockIdx.x * blockDim.x + threadIdx.x;  // 128*1024 threads
  const int row = tid >> 10, col = tid & 1023;
  float v = 0.f, bv = 0.f;
  if (row < 32) {
    const int h = row >> 1, slot = (row & 1) * 2;
    v  = Wq[(size_t)(h * 64 + slot) * 1024 + col];
    bv = bq[h * 64 + slot];
  } else if (row < 64) {
    const int rr = row - 32, h = rr >> 1, slot = (rr & 1) * 2;
    v  = Wk[(size_t)(h * 64 + slot) * 1024 + col];
    bv = bk[h * 64 + slot];
  }
  Wp[tid] = f2bf(v);
  if (col == 0) bp[row] = bv;
}

// ---------------------------------------------------------------------------
// Tetra features: qk_small (4096 x 128 f32; col j<32 = Q, 32..63 = K) ->
// qf/kf (B*NH, S, 32) bf16, channels 0..7 = [tetra(slot0), tetra(slot2)],
// 8..31 = 0 (pad to WMMA K=32). Score scalings folded into qf.
// ---------------------------------------------------------------------------
__global__ void tetra_kernel(const float* __restrict__ qk,
                             __bf16* __restrict__ qf, __bf16* __restrict__ kf) {
  const int tid = blockIdx.x * blockDim.x + threadIdx.x;  // 65536
  const int h = tid & 15;
  const int row = tid >> 4;               // b*2048 + s
  const int b = row >> 11, s = row & 2047;
  const float* p = qk + (size_t)row * 128;
  const float vals[4] = { p[h * 2], p[h * 2 + 1], p[32 + h * 2], p[32 + h * 2 + 1] };
  float ch[4][4];
#pragma unroll
  for (int i = 0; i < 4; ++i) {
    const float a  = 1.f / (1.f + __expf(-vals[i]));
    const float na = 1.f - a;
    float nei = 1.f - (a + na);
    nei = fminf(fmaxf(nei, 0.f), 1.f);
    ch[i][0] = a; ch[i][1] = na; ch[i][2] = a * na; ch[i][3] = nei;
  }
  const size_t base = ((size_t)(b * 16 + h) * 2048 + s) * 32;
#pragma unroll
  for (int c = 0; c < 4; ++c) {
    qf[base + c]     = f2bf(ch[0][c] * 0.125f);  // 1/sqrt(HD)
    qf[base + 4 + c] = f2bf(ch[1][c] * 0.1f);
    kf[base + c]     = f2bf(ch[2][c]);
    kf[base + 4 + c] = f2bf(ch[3][c]);
  }
#pragma unroll
  for (int c = 8; c < 32; ++c) { qf[base + c] = f2bf(0.f); kf[base + c] = f2bf(0.f); }
}

// V (4096 x 1024 f32) -> Vt bf16 (B*NH, 64, 2048) = [bh][d][s]
__global__ void vt_kernel(const float* __restrict__ V, __bf16* __restrict__ Vt) {
  const int tid = blockIdx.x * blockDim.x + threadIdx.x;  // 4096*1024
  const int col = tid & 1023;
  const int row = tid >> 10;              // b*2048 + s
  const int b = row >> 11, s = row & 2047;
  const int h = col >> 6, d = col & 63;
  Vt[((size_t)(b * 16 + h) * 64 + d) * 2048 + s] = f2bf(V[tid]);
}

// ---------------------------------------------------------------------------
// Flash attention. Grid (B*NH=32, S/64=32), block 128 = 4 waves; each wave
// owns a 16-query tile and streams 2048 keys in chunks of 32.
// scores: 2x bf16 WMMA (channels padded 8->32); P via per-wave LDS tile
// (C-layout -> A-fragment layout); P@V: 4x bf16 WMMA. Output written bf16
// in (B,S,H) layout so the projection GEMM reads it directly.
// ---------------------------------------------------------------------------
__global__ __launch_bounds__(128) void attn_kernel(
    const __bf16* __restrict__ qf, const __bf16* __restrict__ kf,
    const __bf16* __restrict__ Vt, __bf16* __restrict__ out) {
  const int lane  = threadIdx.x & 31;
  const int wave  = threadIdx.x >> 5;     // 0..3
  const int lrow  = lane & 15;
  const int lhalf = lane >> 4;
  const int bh    = blockIdx.x;           // 0..31
  const int qbase = blockIdx.y * 64 + wave * 16;

  __shared__ __align__(64) __bf16 P[4][16][32];   // per-wave 16q x 32k bf16

  const v16bf aq =
      *(const v16bf*)(qf + ((size_t)(bh * 2048 + qbase + lrow)) * 32 + lhalf * 16);

  const __bf16* kfb = kf + (size_t)bh * 2048 * 32;
  const __bf16* vtb = Vt + (size_t)bh * 64 * 2048;

  float m[8], l[8];
#pragma unroll
  for (int r = 0; r < 8; ++r) { m[r] = -1e30f; l[r] = 0.f; }
  v8f o[4] = {};

  for (int kc = 0; kc < 2048; kc += 32) {
    if (kc + 32 < 2048) {
      __builtin_prefetch(kfb + (size_t)(kc + 32 + lrow) * 32, 0, 1);
      __builtin_prefetch(vtb + (size_t)lrow * 2048 + kc + 32, 0, 1);
    }
    const v16bf kb0 = *(const v16bf*)(kfb + (size_t)(kc + lrow) * 32 + lhalf * 16);
    const v16bf kb1 = *(const v16bf*)(kfb + (size_t)(kc + 16 + lrow) * 32 + lhalf * 16);
    v8f z0 = {}, z1 = {};
    const v8f s0 = WMMA_BF16(aq, kb0, z0);
    const v8f s1 = WMMA_BF16(aq, kb1, z1);

    // Online softmax. C layout: (row = r + 8*lhalf, key-col = lrow).
#pragma unroll
    for (int r = 0; r < 8; ++r) {
      float v = fmaxf(s0[r], s1[r]);
      v = fmaxf(v, __shfl_xor(v, 1, 32));
      v = fmaxf(v, __shfl_xor(v, 2, 32));
      v = fmaxf(v, __shfl_xor(v, 4, 32));
      v = fmaxf(v, __shfl_xor(v, 8, 32));
      const float mn = fmaxf(m[r], v);
      const float cf = __expf(m[r] - mn);
      m[r] = mn;
      const float p0 = __expf(s0[r] - mn);
      const float p1 = __expf(s1[r] - mn);
      float rs = p0 + p1;
      rs += __shfl_xor(rs, 1, 32);
      rs += __shfl_xor(rs, 2, 32);
      rs += __shfl_xor(rs, 4, 32);
      rs += __shfl_xor(rs, 8, 32);
      l[r] = l[r] * cf + rs;
#pragma unroll
      for (int t = 0; t < 4; ++t) o[t][r] *= cf;
      P[wave][r + 8 * lhalf][lrow]      = f2bf(p0);
      P[wave][r + 8 * lhalf][16 + lrow] = f2bf(p1);
    }

    const v16bf pa = *(const v16bf*)(&P[wave][lrow][lhalf * 16]);
#pragma unroll
    for (int t = 0; t < 4; ++t) {
      const v16bf vb =
          *(const v16bf*)(vtb + (size_t)(t * 16 + lrow) * 2048 + kc + lhalf * 16);
      o[t] = WMMA_BF16(pa, vb, o[t]);
    }
  }

  const int b = bh >> 4, h = bh & 15;
#pragma unroll
  for (int r = 0; r < 8; ++r) {
    const float inv = 1.f / l[r];
    const size_t row = (size_t)(b * 2048 + qbase + r + 8 * lhalf) * 1024;
#pragma unroll
    for (int t = 0; t < 4; ++t)
      out[row + h * 64 + t * 16 + lrow] = f2bf(o[t][r] * inv);
  }
}

// ---------------------------------------------------------------------------
extern "C" void kernel_launch(void* const* d_in, const int* in_sizes, int n_in,
                              void* d_out, int out_size, void* d_ws, size_t ws_size,
                              hipStream_t stream) {
  (void)in_sizes; (void)n_in; (void)out_size; (void)ws_size;
  const float* x  = (const float*)d_in[0];
  const float* Wq = (const float*)d_in[1];
  const float* Wk = (const float*)d_in[2];
  const float* Wv = (const float*)d_in[3];
  const float* Wo = (const float*)d_in[4];
  const float* bq = (const float*)d_in[5];
  const float* bk = (const float*)d_in[6];
  const float* bv = (const float*)d_in[7];
  const float* bo = (const float*)d_in[8];
  float* out = (float*)d_out;

  char* ws = (char*)d_ws;
  size_t off = 0;
  auto wsalloc = [&](size_t bytes) -> char* {
    char* p = ws + off;
    off = (off + bytes + 255) & ~(size_t)255;
    return p;
  };
  __bf16* xb   = (__bf16*)wsalloc((size_t)4096 * 1024 * sizeof(__bf16));
  __bf16* Wvb  = (__bf16*)wsalloc((size_t)1024 * 1024 * sizeof(__bf16));
  __bf16* Wob  = (__bf16*)wsalloc((size_t)1024 * 1024 * sizeof(__bf16));
  __bf16* Wp   = (__bf16*)wsalloc((size_t)128 * 1024 * sizeof(__bf16));
  float*  bp   = (float*)wsalloc(128 * sizeof(float));
  float*  qk   = (float*)wsalloc((size_t)4096 * 128 * sizeof(float));
  float*  V    = (float*)wsalloc((size_t)4096 * 1024 * sizeof(float));
  __bf16* Vt   = (__bf16*)wsalloc((size_t)32 * 64 * 2048 * sizeof(__bf16));
  __bf16* qfp  = (__bf16*)wsalloc((size_t)32 * 2048 * 32 * sizeof(__bf16));
  __bf16* kfp  = (__bf16*)wsalloc((size_t)32 * 2048 * 32 * sizeof(__bf16));
  __bf16* attn = (__bf16*)wsalloc((size_t)4096 * 1024 * sizeof(__bf16));

  cvt_bf16_kernel<<<16384, 256, 0, stream>>>(x, xb, 4096 * 1024);
  cvt_bf16_kernel<<<4096, 256, 0, stream>>>(Wv, Wvb, 1024 * 1024);
  cvt_bf16_kernel<<<4096, 256, 0, stream>>>(Wo, Wob, 1024 * 1024);
  pack_qk_kernel<<<512, 256, 0, stream>>>(Wq, Wk, bq, bk, Wp, bp);

  gemm_bf16_kernel<<<dim3(64, 1), 256, 0, stream>>>(xb, Wp, bp, qk, 4096, 128, 1024);
  gemm_bf16_kernel<<<dim3(64, 8), 256, 0, stream>>>(xb, Wvb, bv, V, 4096, 1024, 1024);
  tetra_kernel<<<256, 256, 0, stream>>>(qk, qfp, kfp);
  vt_kernel<<<16384, 256, 0, stream>>>(V, Vt);
  attn_kernel<<<dim3(32, 32), 128, 0, stream>>>(qfp, kfp, Vt, attn);
  gemm_bf16_kernel<<<dim3(64, 8), 256, 0, stream>>>(attn, Wob, bo, out, 4096, 1024, 1024);
}
